// CMultiHeadAttention_59614146068936
// MI455X (gfx1250) — matrix operand order repaired and checked by reading, not compiled, not run on hardware
//
#include <hip/hip_runtime.h>
#include <hip/hip_bf16.h>
#include <cmath>

typedef __attribute__((ext_vector_type(16))) _Float16 v16h;
typedef __attribute__((ext_vector_type(8)))  _Float16 v8h;
typedef __attribute__((ext_vector_type(8)))  float    v8f;

constexpr int Bn  = 4;
constexpr int Ln  = 2048;
constexpr int Dn  = 512;
constexpr int Hn  = 8;
constexpr int Cn  = 3;
constexpr int DHn = 64;

// ---------------------------------------------------------------------------
// operand packing helpers (ISA 7.12.2 dense f16 layouts)
//   A 16x32: lane<16 -> K {0..7, 16..23}; lane>=16 -> +8  (two contiguous runs)
//   B 32x16: same as A, lane<16 -> K {0..7, 16..23}; lane>=16 -> +8 (two contiguous runs)
// ---------------------------------------------------------------------------
__device__ __forceinline__ v16h pack_a8(const float* __restrict__ g0,
                                        const float* __restrict__ g1) {
  float4 x0 = ((const float4*)g0)[0];
  float4 x1 = ((const float4*)g0)[1];
  float4 y0 = ((const float4*)g1)[0];
  float4 y1 = ((const float4*)g1)[1];
  v16h a;
  a[0]=(_Float16)x0.x;  a[1]=(_Float16)x0.y;  a[2]=(_Float16)x0.z;  a[3]=(_Float16)x0.w;
  a[4]=(_Float16)x1.x;  a[5]=(_Float16)x1.y;  a[6]=(_Float16)x1.z;  a[7]=(_Float16)x1.w;
  a[8]=(_Float16)y0.x;  a[9]=(_Float16)y0.y;  a[10]=(_Float16)y0.z; a[11]=(_Float16)y0.w;
  a[12]=(_Float16)y1.x; a[13]=(_Float16)y1.y; a[14]=(_Float16)y1.z; a[15]=(_Float16)y1.w;
  return a;
}

__device__ __forceinline__ v16h load_b16(const _Float16* __restrict__ p) {
  v8h lo = ((const v8h*)p)[0];
  v8h hi = ((const v8h*)p)[2];
  v16h b;
#pragma unroll
  for (int e = 0; e < 8; ++e) { b[e] = lo[e]; b[8 + e] = hi[e]; }
  return b;
}

// C = A(f32 row-major, lda) x BT(f16, [n][k] transposed, ldbt), 16x16 tile per wave
__device__ __forceinline__ v8f wmma_tile_a32_bt16(const float* __restrict__ A,
                                                  const _Float16* __restrict__ BT,
                                                  int lda, int ldbt,
                                                  int m0, int n0, int K, int lane) {
  v8f acc = {};
  const float*    ar = A  + (size_t)(m0 + (lane & 15)) * lda  + ((lane >= 16) ? 8 : 0);
  const _Float16* br = BT + (size_t)(n0 + (lane & 15)) * ldbt + ((lane >= 16) ? 8 : 0);
  for (int k0 = 0; k0 < K; k0 += 32) {
    if (k0 + 128 < K) {
      __builtin_prefetch(ar + k0 + 128, 0, 1);
      __builtin_prefetch(br + k0 + 128, 0, 1);
    }
    v16h a = pack_a8(ar + k0, ar + k0 + 16);
    v16h b = load_b16(br + k0);
    acc = __builtin_amdgcn_wmma_f32_16x16x32_f16(false, a, false, b,
                                                 (short)0, acc, false, false);
  }
  return acc;
}

// ---------------------------------------------------------------------------
// weight transpose + f16 convert: WT[n][k] = (f16) W[k][n], for Wv and Wo
// ---------------------------------------------------------------------------
__global__ void wconv_kernel(const float* __restrict__ Wv, const float* __restrict__ Wo,
                             _Float16* __restrict__ WvT, _Float16* __restrict__ WoT) {
  int t = blockIdx.x * blockDim.x + threadIdx.x;
  const int nEl = Dn * Dn;
  if (t >= 2 * nEl) return;
  bool second = (t >= nEl);
  int i  = second ? t - nEl : t;
  int kk = i / Dn, nn = i % Dn;
  const float* W = second ? Wo : Wv;
  _Float16*   WT = second ? WoT : WvT;
  WT[(size_t)nn * Dn + kk] = (_Float16)W[i];
}

// ---------------------------------------------------------------------------
// qa/ka projections: qa[b,h,l] = q[b,l,:] . Wq[:,h] + bq[h]
// ---------------------------------------------------------------------------
__global__ void proj_qk_kernel(const float* __restrict__ q, const float* __restrict__ k,
                               const float* __restrict__ Wq, const float* __restrict__ bq,
                               const float* __restrict__ Wk, const float* __restrict__ bk,
                               float* __restrict__ qa, float* __restrict__ ka) {
  int t = blockIdx.x * blockDim.x + threadIdx.x;
  const int n = Bn * Ln * Hn;
  if (t >= 2 * n) return;
  bool isK = (t >= n);
  int idx = isK ? (t - n) : t;
  int h  = idx % Hn;
  int l  = (idx / Hn) % Ln;
  int bi = idx / (Hn * Ln);
  const float* X    = isK ? k  : q;
  const float* W    = isK ? Wk : Wq;
  const float* bias = isK ? bk : bq;
  const float* row  = X + ((size_t)bi * Ln + l) * Dn;
  float acc = bias[h];
  for (int d = 0; d < Dn; ++d) acc = fmaf(row[d], W[d * Hn + h], acc);
  float* dst = isK ? ka : qa;
  dst[((size_t)bi * Hn + h) * Ln + l] = acc;
}

// ---------------------------------------------------------------------------
// vhT = transpose_to[b][h][dh][l] of (v @ Wv + bv), stored f16.
// one wave per 16x16 tile of the 8192x512 GEMM.
// ---------------------------------------------------------------------------
__global__ void vproj_kernel(const float* __restrict__ v, const _Float16* __restrict__ WvT,
                             const float* __restrict__ bv, _Float16* __restrict__ vhT) {
  int wave = (blockIdx.x * blockDim.x + threadIdx.x) >> 5;
  int lane = threadIdx.x & 31;
  const int tilesN = Dn / 16;                 // 32
  int m0 = (wave / tilesN) * 16;
  int n0 = (wave % tilesN) * 16;
  if (m0 >= Bn * Ln) return;
  v8f acc = wmma_tile_a32_bt16(v, WvT, Dn, Dn, m0, n0, Dn, lane);
  int nc = n0 + (lane & 15);
  int h = nc >> 6, dh = nc & 63;
  int rbase = (lane >= 16) ? 8 : 0;
  float bias = bv[nc];
  int m  = m0 + rbase;
  int bi = m / Ln, l = m % Ln;
  _Float16* dst = vhT + (((size_t)bi * Hn + h) * DHn + dh) * Ln + l;
#pragma unroll
  for (int r = 0; r < 8; ++r) dst[r] = (_Float16)(acc[r] + bias);
}

// ---------------------------------------------------------------------------
// main attention: one block (256 thr = 8 waves) per (b, h, 16-row i-tile)
// ---------------------------------------------------------------------------
__global__ void attn_kernel(const float* __restrict__ qa, const float* __restrict__ ka,
                            const _Float16* __restrict__ vhT,
                            const float* __restrict__ d_q, const float* __restrict__ d_k_top,
                            const float* __restrict__ d_k_bot, const float* __restrict__ d_k_score,
                            const float* __restrict__ Ww, const float* __restrict__ bw,
                            const float* __restrict__ Wb, const float* __restrict__ bbv,
                            float* __restrict__ probs, float* __restrict__ attn) {
  extern __shared__ float smem[];
  float* sS   = smem;                  // 16*2048 scores / probs
  float* sRed = sS   + 16 * Ln;        // 1024  cross-wave reduction
  float* sKa  = sRed + 1024;           // 512   staged ka chunk
  float* sT   = sKa  + 512;            // 1536  staged d_k_top chunk [512][3]
  float* sBt  = sT   + 1536;           // 1536  staged d_k_bot chunk
  float* sSc  = sBt  + 1536;           // 3072  staged d_k_score chunk [512][2][3]
  float* sQa  = sSc  + 3072;           // 16
  float* sDq  = sQa  + 16;             // 48

  int bid = blockIdx.x;
  int it  = bid & 127;                 // L/16 = 128 tiles
  int h   = (bid >> 7) & 7;
  int bi  = bid >> 10;
  int i0  = it * 16;
  int tid = threadIdx.x;
  int lane = tid & 31, wave = tid >> 5;

  if (tid < 16) sQa[tid] = qa[((size_t)bi * Hn + h) * Ln + i0 + tid];
  if (tid >= 32 && tid < 32 + 48) {
    int r = (tid - 32) / 3, c = (tid - 32) % 3;
    sDq[r * 3 + c] = d_q[((size_t)bi * Ln + i0 + r) * Cn + c];
  }
  float ww0 = Ww[0 * Hn + h], ww1 = Ww[1 * Hn + h], ww2 = Ww[2 * Hn + h];
  float wb0 = Wb[0 * Hn + h], wb1 = Wb[1 * Hn + h], wb2 = Wb[2 * Hn + h];
  float bwh = bw[h], bbh = bbv[h];

  // ---- phase 1: scores, 512-j chunks staged through LDS ----
  for (int j0 = 0; j0 < Ln; j0 += 512) {
    __syncthreads();                   // protect previous chunk / init stage
    {
      const float4* g = (const float4*)(ka + ((size_t)bi * Hn + h) * Ln + j0);
      float4* s = (float4*)sKa;
      for (int t = tid; t < 128; t += 256) s[t] = g[t];
      g = (const float4*)(d_k_top + ((size_t)bi * Ln + j0) * 3);
      s = (float4*)sT;
      for (int t = tid; t < 384; t += 256) s[t] = g[t];
      g = (const float4*)(d_k_bot + ((size_t)bi * Ln + j0) * 3);
      s = (float4*)sBt;
      for (int t = tid; t < 384; t += 256) s[t] = g[t];
      g = (const float4*)(d_k_score + ((size_t)bi * Ln + j0) * 6);
      s = (float4*)sSc;
      for (int t = tid; t < 768; t += 256) s[t] = g[t];
    }
    __syncthreads();
    for (int idx = tid; idx < 16 * 512; idx += 256) {
      int row = idx >> 9, jj = idx & 511;
      float dq0 = sDq[row * 3 + 0], dq1 = sDq[row * 3 + 1], dq2 = sDq[row * 3 + 2];
      float dm0 = fabsf(dq0 - sT[jj*3+0]) * sSc[jj*6+0] + fabsf(dq0 - sBt[jj*3+0]) * sSc[jj*6+3];
      float dm1 = fabsf(dq1 - sT[jj*3+1]) * sSc[jj*6+1] + fabsf(dq1 - sBt[jj*3+1]) * sSc[jj*6+4];
      float dm2 = fabsf(dq2 - sT[jj*3+2]) * sSc[jj*6+2] + fabsf(dq2 - sBt[jj*3+2]) * sSc[jj*6+5];
      float x   = fmaf(dm0, ww0, fmaf(dm1, ww1, fmaf(dm2, ww2, bwh)));
      float sp  = fmaxf(x, 0.f) + __logf(1.f + __expf(-fabsf(x)));  // stable softplus
      float db  = fmaf(dm0, wb0, fmaf(dm1, wb1, fmaf(dm2, wb2, bbh)));
      sS[row * Ln + j0 + jj] = sQa[row] * sKa[jj] * sp + db;
    }
  }
  __syncthreads();

  // ---- phase 2: exact softmax per row (one wave per 2 rows), probs -> global ----
  for (int rr = 0; rr < 2; ++rr) {
    int row = wave * 2 + rr;
    float* srow = sS + row * Ln;
    float m = -INFINITY;
    for (int j = lane * 4; j < Ln; j += 128) {
      float4 sv = *(const float4*)&srow[j];
      m = fmaxf(m, fmaxf(fmaxf(sv.x, sv.y), fmaxf(sv.z, sv.w)));
    }
    for (int off = 16; off > 0; off >>= 1) m = fmaxf(m, __shfl_xor(m, off, 32));
    float sum = 0.f;
    for (int j = lane * 4; j < Ln; j += 128) {
      float4 sv = *(float4*)&srow[j];
      sv.x = __expf(sv.x - m); sv.y = __expf(sv.y - m);
      sv.z = __expf(sv.z - m); sv.w = __expf(sv.w - m);
      sum += (sv.x + sv.y) + (sv.z + sv.w);
      *(float4*)&srow[j] = sv;
    }
    for (int off = 16; off > 0; off >>= 1) sum += __shfl_xor(sum, off, 32);
    float inv = 1.f / sum;
    float* prow = probs + (((size_t)bi * Hn + h) * Ln + i0 + row) * (size_t)Ln;
    for (int j = lane * 4; j < Ln; j += 128) {
      float4 e = *(float4*)&srow[j];
      e.x *= inv; e.y *= inv; e.z *= inv; e.w *= inv;
      *(float4*)&srow[j] = e;
      *(float4*)&prow[j] = e;
    }
  }
  __syncthreads();

  // ---- phase 3: out16x64 = probs(16x2048) @ vh(2048x64), WMMA, 4 N-tiles x 2 K-halves ----
  int nt = wave & 3;
  int kh = wave >> 2;
  const float* aBase = sS + (size_t)(lane & 15) * Ln + ((lane >= 16) ? 8 : 0);
  const _Float16* bBase = vhT
      + (((size_t)bi * Hn + h) * DHn + nt * 16 + (lane & 15)) * Ln
      + ((lane >= 16) ? 8 : 0);
  v8f acc = {};
  for (int k0 = kh * 1024; k0 < kh * 1024 + 1024; k0 += 32) {
    v16h a = pack_a8(aBase + k0, aBase + k0 + 16);
    v16h b = load_b16(bBase + k0);
    acc = __builtin_amdgcn_wmma_f32_16x16x32_f16(false, a, false, b,
                                                 (short)0, acc, false, false);
  }
  if (kh == 1) {
#pragma unroll
    for (int r = 0; r < 8; ++r) sRed[nt * 256 + r * 32 + lane] = acc[r];
  }
  __syncthreads();
  if (kh == 0) {
    int nc = nt * 16 + (lane & 15);
    int rbase = (lane >= 16) ? 8 : 0;
#pragma unroll
    for (int r = 0; r < 8; ++r) {
      float val = acc[r] + sRed[nt * 256 + r * 32 + lane];
      int gi = i0 + rbase + r;
      attn[((size_t)bi * Ln + gi) * Dn + h * DHn + nc] = val;
    }
  }
}

// ---------------------------------------------------------------------------
// out = attn @ Wo + bo  (8192x512x512 WMMA GEMM, WoT pre-transposed f16)
// ---------------------------------------------------------------------------
__global__ void oproj_kernel(const float* __restrict__ attn, const _Float16* __restrict__ WoT,
                             const float* __restrict__ bo, float* __restrict__ out) {
  int wave = (blockIdx.x * blockDim.x + threadIdx.x) >> 5;
  int lane = threadIdx.x & 31;
  const int tilesN = Dn / 16;
  int m0 = (wave / tilesN) * 16;
  int n0 = (wave % tilesN) * 16;
  if (m0 >= Bn * Ln) return;
  v8f acc = wmma_tile_a32_bt16(attn, WoT, Dn, Dn, m0, n0, Dn, lane);
  int nc = n0 + (lane & 15);
  float bias = bo[nc];
  int rbase = (lane >= 16) ? 8 : 0;
#pragma unroll
  for (int r = 0; r < 8; ++r)
    out[(size_t)(m0 + rbase + r) * Dn + nc] = acc[r] + bias;
}

// ---------------------------------------------------------------------------
extern "C" void kernel_launch(void* const* d_in, const int* in_sizes, int n_in,
                              void* d_out, int out_size, void* d_ws, size_t ws_size,
                              hipStream_t stream) {
  const float* q        = (const float*)d_in[0];
  const float* k        = (const float*)d_in[1];
  const float* v        = (const float*)d_in[2];
  const float* d_q      = (const float*)d_in[3];
  const float* d_k_top  = (const float*)d_in[4];
  const float* d_k_bot  = (const float*)d_in[5];
  const float* d_k_sc   = (const float*)d_in[6];
  const float* Wq       = (const float*)d_in[7];
  const float* bq       = (const float*)d_in[8];
  const float* Wk       = (const float*)d_in[9];
  const float* bk       = (const float*)d_in[10];
  // d_in[11]=Wc, d_in[12]=bc : unused by the reference
  const float* Wv       = (const float*)d_in[13];
  const float* bv       = (const float*)d_in[14];
  const float* Wo       = (const float*)d_in[15];
  const float* bo       = (const float*)d_in[16];
  const float* Ww       = (const float*)d_in[17];
  const float* bw       = (const float*)d_in[18];
  const float* Wb       = (const float*)d_in[19];
  const float* bb       = (const float*)d_in[20];

  float* ws    = (float*)d_ws;
  float* qa    = ws;                                        // 65,536 f32
  float* ka    = qa + (size_t)Bn * Hn * Ln;                 // 65,536 f32
  float* attn  = ka + (size_t)Bn * Hn * Ln;                 // 4,194,304 f32
  _Float16* vhT = (_Float16*)(attn + (size_t)Bn * Ln * Dn); // 4,194,304 f16
  _Float16* WvT = vhT + (size_t)Bn * Hn * DHn * Ln;         // 262,144 f16
  _Float16* WoT = WvT + (size_t)Dn * Dn;                    // 262,144 f16

  float* out   = (float*)d_out;
  float* probs = out + (size_t)Bn * Ln * Dn;

  // 0) transpose+convert weights to f16 [n][k]
  wconv_kernel<<<(2 * Dn * Dn + 255) / 256, 256, 0, stream>>>(Wv, Wo, WvT, WoT);
  // 1) qa / ka scalar projections
  {
    int total = 2 * Bn * Ln * Hn;
    proj_qk_kernel<<<(total + 255) / 256, 256, 0, stream>>>(q, k, Wq, bq, Wk, bk, qa, ka);
  }
  // 2) vhT = (v @ Wv + bv)^T per head, f16
  {
    int waves = (Bn * Ln / 16) * (Dn / 16);                 // 16384
    vproj_kernel<<<waves / 8, 256, 0, stream>>>(v, WvT, bv, vhT);
  }
  // 3) scores + softmax + probs + probs@vh
  {
    size_t lds = (size_t)(16 * Ln + 1024 + 512 + 1536 + 1536 + 3072 + 16 + 48) * sizeof(float);
    attn_kernel<<<Bn * Hn * (Ln / 16), 256, lds, stream>>>(
        qa, ka, vhT, d_q, d_k_top, d_k_bot, d_k_sc, Ww, bw, Wb, bb, probs, attn);
  }
  // 4) out = attn @ Wo + bo
  {
    int waves = (Bn * Ln / 16) * (Dn / 16);
    oproj_kernel<<<waves / 8, 256, 0, stream>>>(attn, WoT, bo, out);
  }
}
